// Evaluation_90202903150671
// MI455X (gfx1250) — matrix operand order repaired and checked by reading, not compiled, not run on hardware
//
#include <hip/hip_runtime.h>

typedef __attribute__((ext_vector_type(16))) _Float16 v16h;
typedef __attribute__((ext_vector_type(8)))  float    v8f;

// Problem dims (fixed by the harness / setup_inputs)
constexpr int B_ = 2, C_ = 32, H_ = 128, W_ = 160, D_ = 48, S_ = 2, G_ = 8;
constexpr int HW = H_ * W_;

// Stage-3 intrinsics hardcoded in the reference
__device__ __forceinline__ void compute_rt(const float* __restrict__ src_proj,
                                           const float* __restrict__ ref_proj,
                                           float R[9], float T[3]) {
  const float KI[9] = {0.00276594f, 0.f, -0.2846162f,
                       0.f, 0.00277472f, -0.21471854f,
                       0.f, 0.f, 1.f};
  const float K_[9] = {361.54126f, 0.f, 102.9005f,
                       0.f, 360.39624f, 77.38375f,
                       0.f, 0.f, 1.f};
  float sp[12], rp[12];
#pragma unroll
  for (int i = 0; i < 3; ++i)
#pragma unroll
    for (int j = 0; j < 4; ++j) {
      float a = 0.f, b = 0.f;
#pragma unroll
      for (int k = 0; k < 3; ++k) {
        a += KI[i * 3 + k] * src_proj[k * 4 + j];
        b += KI[i * 3 + k] * ref_proj[k * 4 + j];
      }
      sp[i * 4 + j] = a;
      rp[i * 4 + j] = b;
    }
  float P[9];
#pragma unroll
  for (int i = 0; i < 3; ++i)
#pragma unroll
    for (int k = 0; k < 3; ++k) {
      float a = 0.f;
#pragma unroll
      for (int j = 0; j < 3; ++j) a += sp[i * 4 + j] * rp[k * 4 + j];
      P[i * 3 + k] = a;  // R_src @ R_ref^T
    }
  float tv[3];
#pragma unroll
  for (int i = 0; i < 3; ++i) {
    float a = 0.f;
#pragma unroll
    for (int k = 0; k < 3; ++k) a += P[i * 3 + k] * rp[k * 4 + 3];
    tv[i] = sp[i * 4 + 3] - a;
  }
#pragma unroll
  for (int i = 0; i < 3; ++i) {
    float a = 0.f;
#pragma unroll
    for (int k = 0; k < 3; ++k) a += K_[i * 3 + k] * tv[k];
    T[i] = a;  // trans = INTR @ (t_src - P t_ref)
  }
  float KP[9];
#pragma unroll
  for (int i = 0; i < 3; ++i)
#pragma unroll
    for (int j = 0; j < 3; ++j) {
      float a = 0.f;
#pragma unroll
      for (int k = 0; k < 3; ++k) a += K_[i * 3 + k] * P[k * 3 + j];
      KP[i * 3 + j] = a;
    }
#pragma unroll
  for (int i = 0; i < 3; ++i)
#pragma unroll
    for (int j = 0; j < 3; ++j) {
      float a = 0.f;
#pragma unroll
      for (int k = 0; k < 3; ++k) a += KP[i * 3 + k] * KI[k * 3 + j];
      R[i * 3 + j] = a;  // rot = INTR @ P @ INTR_INV
    }
}

// One wave per (b, hw) pixel.
// A tile [16 depths x 32 channels] of warped features (f16),
// B      [32 channels x 16 cols]  block-diagonal ref * 0.25 (f16, cols 8..15 zero),
// D = A@B + C accumulated over S views with v_wmma_f32_16x16x32_f16.
//
// Cache policy: src_features gathers keep default RT (the only reused working
// set, ~5.2 MB/view, must stay hot in WGP$/L2); depth/ref loads and the 63 MB
// output stream are one-touch -> non-temporal so they don't evict it.
__global__ __launch_bounds__(256) void gwc_warp_wmma_kernel(
    const float* __restrict__ ref_feature,   // [B,C,H,W]
    const float* __restrict__ src_features,  // [S,B,C,H,W]
    const float* __restrict__ ref_proj,      // [B,4,4]
    const float* __restrict__ src_projs,     // [S,B,4,4]
    const float* __restrict__ depth,         // [B,D,H,W]
    float* __restrict__ out)                 // [B,G,D,H,W]
{
  const int lane = threadIdx.x & 31;
  const int wid  = threadIdx.x >> 5;
  const int task = blockIdx.x * 8 + wid;      // grid sized exactly: B*HW/8 blocks
  const int b    = task / HW;
  const int hw   = task - b * HW;
  const int h    = hw / W_;
  const int w    = hw - h * W_;
  const int col  = lane & 15;                 // A row (depth-in-tile) / B-D column (group)
  const int hi   = lane >> 4;                 // 0: channels {0-7,16-23}, 1: {8-15,24-31}

  // ---- Build B once per pixel: B[k, col] = ref[b, k, hw] * 0.25 iff k//4 == col ----
  const float* refb = ref_feature + (size_t)b * C_ * HW + hw;
  v16h bm;
#pragma unroll
  for (int j = 0; j < 16; ++j) {
    int k = (j & 7) + ((j >> 3) << 4) + hi * 8; // B rows: lanes 0-15 K={0-7,16-23}, 16-31 K={8-15,24-31}
    float rv = __builtin_nontemporal_load(&refb[(size_t)k * HW]);
    bool use = (col < G_) && ((k >> 2) == col);
    bm[j] = (_Float16)(use ? rv * 0.25f : 0.0f);
  }

  v8f acc[3];
#pragma unroll
  for (int t = 0; t < 3; ++t) acc[t] = (v8f){};

  const float wf = (float)w, hf = (float)h;
  const float* dep = depth + (size_t)b * D_ * HW + hw;

  for (int s = 0; s < S_; ++s) {
    float R[9], T[3];
    compute_rt(src_projs + (size_t)(s * B_ + b) * 16,
               ref_proj + (size_t)b * 16, R, T);
    const float rx = R[0] * wf + R[1] * hf + R[2];
    const float ry = R[3] * wf + R[4] * hf + R[5];
    const float rz = R[6] * wf + R[7] * hf + R[8];
    // channel-half offset folded into the base pointer; per-channel offsets are immediates
    const float* sbase = src_features + (size_t)(s * B_ + b) * C_ * HW + (size_t)hi * 8 * HW;

#pragma unroll
    for (int t = 0; t < 3; ++t) {
      const int d = t * 16 + col;             // depth handled by this lane-pair (A row)
      const float dv = __builtin_nontemporal_load(&dep[(size_t)d * HW]);
      float px = rx * dv + T[0];
      float py = ry * dv + T[1];
      float pz = rz * dv + T[2];
      const bool zok = pz > 0.001f;
      px = zok ? px : 0.0f;
      py = zok ? py : 0.0f;
      pz = zok ? pz : 1.0f;
      float fx = px / pz;
      float fy = py / pz;
      fx = (fx >= 0.0f && fx < (float)W_) ? fx : 0.0f;
      fy = (fy >= 0.0f && fy < (float)H_) ? fy : 0.0f;

      const float x0f = floorf(fx), y0f = floorf(fy);
      const float ax = fx - x0f, ay = fy - y0f;
      const float wa = (1.0f - ax) * (1.0f - ay);
      const float wb = (1.0f - ax) * ay;
      const float wc = ax * (1.0f - ay);
      const float wd = ax * ay;
      const int x0 = (int)x0f, y0 = (int)y0f;
      const int x1 = (x0 + 1 == W_) ? 0 : x0 + 1;   // reference wraps modulo W/H
      const int y1 = (y0 + 1 == H_) ? 0 : y0 + 1;
      const int ia  = y0 * W_ + x0;
      const int ib2 = y1 * W_ + x0;
      const int ic  = y0 * W_ + x1;
      const int id2 = y1 * W_ + x1;

      v16h am;
#pragma unroll
      for (int j = 0; j < 16; ++j) {
        // ISA 16-bit A layout: slots 0..7 -> K 0..7, slots 8..15 -> K 16..23 (low half),
        // +8 for lanes 16..31 (folded into sbase).
        const int c = (j & 7) + ((j >> 3) << 4);
        const float* p = sbase + (size_t)c * HW;
        const float v = wa * p[ia] + wb * p[ib2] + wc * p[ic] + wd * p[id2];
        am[j] = (_Float16)v;
      }
      acc[t] = __builtin_amdgcn_wmma_f32_16x16x32_f16(
          /*neg_a=*/false, am, /*neg_b=*/false, bm,
          /*c_mod=*/(short)0, acc[t], /*reuse_a=*/false, /*reuse_b=*/false);
    }
  }

  // ---- Store: D layout f32 16x16 — lane col = group, VGPR r = depth r (+8 for hi lanes) ----
  if (col < G_) {
    float* ob = out + ((size_t)(b * G_ + col) * D_ + hi * 8) * HW + hw;
#pragma unroll
    for (int t = 0; t < 3; ++t)
#pragma unroll
      for (int r = 0; r < 8; ++r)
        __builtin_nontemporal_store(acc[t][r], &ob[(size_t)(t * 16 + r) * HW]);
  }
}

extern "C" void kernel_launch(void* const* d_in, const int* in_sizes, int n_in,
                              void* d_out, int out_size, void* d_ws, size_t ws_size,
                              hipStream_t stream) {
  (void)in_sizes; (void)n_in; (void)out_size; (void)d_ws; (void)ws_size;
  const float* ref_feature = (const float*)d_in[0];
  const float* src_features = (const float*)d_in[1];
  const float* ref_proj = (const float*)d_in[2];
  const float* src_projs = (const float*)d_in[3];
  const float* depth_sample = (const float*)d_in[4];
  float* out = (float*)d_out;

  const int waves = B_ * HW;            // one wave per (b, pixel)
  const int blocks = waves / 8;         // 8 waves (256 threads) per block; divides exactly
  gwc_warp_wmma_kernel<<<blocks, 256, 0, stream>>>(
      ref_feature, src_features, ref_proj, src_projs, depth_sample, out);
}
